// HeteroGNNLinkPredictor_66348654788681
// MI455X (gfx1250) — matrix operand order repaired and checked by reading, not compiled, not run on hardware
//
#include <hip/hip_runtime.h>
#include <hip/hip_bf16.h>

// ---------------------------------------------------------------------------
// HeteroGNN (2-layer GAT, 2 edge types) for gfx1250.
//  - src projection hs = x @ Ws via v_wmma_f32_16x16x32_bf16, al_s fused in
//    the epilogue (shuffle reduce).
//  - dst attention logits al_d = x . (Wd @ a_d)  (GEMM removed algebraically)
//  - segment softmax: atomicMax(int-encoded float) pass + fused exp/scatter
//    pass, normalized in a node-wise finalize (+bias, ELU).
// ---------------------------------------------------------------------------

typedef __attribute__((ext_vector_type(16))) __bf16 v16bf;
typedef __attribute__((ext_vector_type(8)))  float  v8f;

// native f32 -> bf16 (fptrunc, RNE) — lets the backend use packed cvt ops
__device__ __forceinline__ __bf16 cvt_bf16(float f) { return (__bf16)f; }

// monotone float <-> int encoding so signed atomicMax orders like float max
__device__ __forceinline__ int   encf(float f) {
  int b = __float_as_int(f);
  return b >= 0 ? b : (b ^ 0x7fffffff);
}
__device__ __forceinline__ float decf(int e) {
  return __int_as_float(e >= 0 ? e : (e ^ 0x7fffffff));
}
#define ENC_NEG_INF ((int)0x807FFFFF)

// --------------------------- init per-conv state ---------------------------
__global__ void __launch_bounds__(256)
init_dst(int* __restrict__ mEnc, float* __restrict__ den,
         float* __restrict__ out, int n_dst) {
  int i = blockIdx.x * blockDim.x + threadIdx.x;
  if (i >= n_dst * 64) return;
  out[i] = 0.f;
  if ((i & 63) == 0) {
    den[i >> 6]  = 0.f;
    mEnc[i >> 6] = ENC_NEG_INF;
  }
}

// --------------------------- wvec = Wd @ a_d (64) --------------------------
__global__ void matvec64(const float* __restrict__ W, const float* __restrict__ a,
                         float* __restrict__ wv) {
  int k = threadIdx.x;
  if (k < 64) {
    float s = 0.f;
    #pragma unroll 8
    for (int c = 0; c < 64; ++c) s += W[k * 64 + c] * a[c];
    wv[k] = s;
  }
}

// --------------------------- al_d[n] = x[n,:].wvec -------------------------
__global__ void __launch_bounds__(256)
rowdot64(const float* __restrict__ X, const float* __restrict__ w,
         float* __restrict__ al, int N) {
  int g = blockIdx.x * blockDim.x + threadIdx.x;
  int row = g >> 5, lane = g & 31;
  if (row >= N) return;
  const float* xr = X + (size_t)row * 64;
  float v = xr[lane] * w[lane] + xr[lane + 32] * w[lane + 32];
  #pragma unroll
  for (int m = 1; m < 32; m <<= 1) v += __shfl_xor(v, m, 32);
  if (lane == 0) al[row] = v;
}

// ------------- hs = X @ Ws (WMMA bf16, f32 acc), al_s fused ----------------
// one wave computes a 16-row x 64-col strip; block = 8 waves = 128 rows.
__global__ void __launch_bounds__(256)
gemm_hs_al(const float* __restrict__ X, const float* __restrict__ Wg,
           const float* __restrict__ avec, float* __restrict__ H,
           float* __restrict__ al, int N) {
  __shared__ __align__(32) __bf16 Wt[64 * 64];   // transposed: Wt[c*64 + k]
  for (int e = threadIdx.x; e < 4096; e += 256) {
    int k = e >> 6, c = e & 63;
    Wt[c * 64 + k] = cvt_bf16(Wg[e]);            // Wg is [k][c] row-major
  }
  __syncthreads();

  int wave = threadIdx.x >> 5;
  int lane = threadIdx.x & 31;
  int tile = blockIdx.x * 8 + wave;
  int row0 = tile * 16;
  if (row0 >= N) return;

  int half = lane >> 4;        // 0: lanes 0-15, 1: lanes 16-31
  int l15  = lane & 15;
  const float4* xr4 = (const float4*)(X + ((size_t)row0 + l15) * 64);

  v8f acc[4] = {v8f{}, v8f{}, v8f{}, v8f{}};

  #pragma unroll
  for (int k0 = 0; k0 < 64; k0 += 32) {
    // A fragment per ISA 16-bit 16x32 layout:
    //  lanes 0-15 : elems 0-7 -> K=k0+0..7,  elems 8-15 -> K=k0+16..23
    //  lanes 16-31: elems 0-7 -> K=k0+8..15, elems 8-15 -> K=k0+24..31
    int kb = k0 + half * 8;                     // multiple of 8 -> 16B aligned
    float4 f0 = xr4[(kb >> 2) + 0];
    float4 f1 = xr4[(kb >> 2) + 1];
    float4 f2 = xr4[(kb >> 2) + 4];             // kb + 16
    float4 f3 = xr4[(kb >> 2) + 5];
    v16bf afrag;
    afrag[0]  = cvt_bf16(f0.x); afrag[1]  = cvt_bf16(f0.y);
    afrag[2]  = cvt_bf16(f0.z); afrag[3]  = cvt_bf16(f0.w);
    afrag[4]  = cvt_bf16(f1.x); afrag[5]  = cvt_bf16(f1.y);
    afrag[6]  = cvt_bf16(f1.z); afrag[7]  = cvt_bf16(f1.w);
    afrag[8]  = cvt_bf16(f2.x); afrag[9]  = cvt_bf16(f2.y);
    afrag[10] = cvt_bf16(f2.z); afrag[11] = cvt_bf16(f2.w);
    afrag[12] = cvt_bf16(f3.x); afrag[13] = cvt_bf16(f3.y);
    afrag[14] = cvt_bf16(f3.z); afrag[15] = cvt_bf16(f3.w);
    #pragma unroll
    for (int ct = 0; ct < 4; ++ct) {
      // B fragment: lane holds col c, elems 0-7 K=k0+half*8..+7, elems 8-15 K=k0+16+half*8..+7
      int c = ct * 16 + l15;
      typedef __attribute__((ext_vector_type(8))) __bf16 v8bf_t;
      v8bf_t blo = *(const v8bf_t*)&Wt[c * 64 + k0 + half * 8];
      v8bf_t bhi = *(const v8bf_t*)&Wt[c * 64 + k0 + 16 + half * 8];
      v16bf bfrag = __builtin_shufflevector(blo, bhi, 0, 1, 2, 3, 4, 5, 6, 7, 8, 9, 10, 11, 12, 13, 14, 15);
      acc[ct] = __builtin_amdgcn_wmma_f32_16x16x32_bf16(
          false, afrag, false, bfrag, (short)0, acc[ct], false, false);
    }
  }

  // epilogue: store hs rows + fused al_s = hs . avec
  int rbase = half * 8;        // C layout: lanes 0-15 rows 0-7, 16-31 rows 8-15
  float p[8] = {0.f, 0.f, 0.f, 0.f, 0.f, 0.f, 0.f, 0.f};
  #pragma unroll
  for (int ct = 0; ct < 4; ++ct) {
    int c = ct * 16 + l15;
    float av = avec[c];
    #pragma unroll
    for (int r = 0; r < 8; ++r) {
      float v = acc[ct][r];
      H[((size_t)row0 + rbase + r) * 64 + c] = v;
      p[r] += v * av;
    }
  }
  #pragma unroll
  for (int m = 1; m < 16; m <<= 1)
    #pragma unroll
    for (int r = 0; r < 8; ++r) p[r] += __shfl_xor(p[r], m, 16);
  if (l15 == 0) {
    #pragma unroll
    for (int r = 0; r < 8; ++r) al[row0 + rbase + r] = p[r];
  }
}

// --------------------------- segment max over edges ------------------------
__global__ void __launch_bounds__(256)
edge_max(const int* __restrict__ src, const int* __restrict__ dst,
         const float* __restrict__ als, const float* __restrict__ ald,
         int* __restrict__ mEnc, int E) {
  int i = blockIdx.x * blockDim.x + threadIdx.x;
  if (i >= E) return;
  int d = dst[i];
  float e = als[src[i]] + ald[d];
  e = e > 0.f ? e : 0.2f * e;                      // leaky_relu(0.2)
  atomicMax(&mEnc[d], encf(e));
}

// --------------- fused exp / denominator / message scatter -----------------
// one wave (32 lanes) per edge; lane covers channels {lane, lane+32}
__global__ void __launch_bounds__(256)
edge_scatter(const int* __restrict__ src, const int* __restrict__ dst,
             const float* __restrict__ als, const float* __restrict__ ald,
             const int* __restrict__ mEnc, const float* __restrict__ hs,
             float* __restrict__ den, float* __restrict__ out, int E) {
  int g = blockIdx.x * blockDim.x + threadIdx.x;
  int e = g >> 5, lane = g & 31;
  if (e >= E) return;
  int s = src[e], d = dst[e];
  float lg = als[s] + ald[d];
  lg = lg > 0.f ? lg : 0.2f * lg;
  float ex = __expf(lg - decf(mEnc[d]));          // <= 1, no overflow
  if (lane == 0) atomicAdd(&den[d], ex);
  const float* hr = hs + (size_t)s * 64;
  float* orow = out + (size_t)d * 64;
  atomicAdd(&orow[lane],      ex * hr[lane]);
  atomicAdd(&orow[lane + 32], ex * hr[lane + 32]);
}

// ---------------- normalize, +bias, ELU, write next features ---------------
__global__ void __launch_bounds__(256)
finalize(const float* __restrict__ acc, const float* __restrict__ den,
         const float* __restrict__ bias, float* __restrict__ o, int n) {
  int i = blockIdx.x * blockDim.x + threadIdx.x;
  if (i >= n * 64) return;
  int node = i >> 6, c = i & 63;
  float dn = den[node];
  float v = (dn > 0.f ? acc[i] / dn : 0.f) + bias[c];
  v = v > 0.f ? v : expm1f(v);                     // ELU
  o[i] = v;
}

// ---------------------------------------------------------------------------
struct Scratch {
  float* hs; float* out; float* als; float* ald; float* den; int* mEnc; float* wv;
};

static void run_conv(const float* x_src, const float* x_dst, int n_src, int n_dst,
                     const int* src, const int* dst, int E,
                     const float* Ws, const float* Wd,
                     const float* a_s, const float* a_d, const float* b,
                     float* h_out, const Scratch& S, hipStream_t st) {
  init_dst<<<(n_dst * 64 + 255) / 256, 256, 0, st>>>(S.mEnc, S.den, S.out, n_dst);
  matvec64<<<1, 64, 0, st>>>(Wd, a_d, S.wv);
  rowdot64<<<((size_t)n_dst * 32 + 255) / 256, 256, 0, st>>>(x_dst, S.wv, S.ald, n_dst);
  gemm_hs_al<<<((n_src + 15) / 16 + 7) / 8, 256, 0, st>>>(x_src, Ws, a_s, S.hs, S.als, n_src);
  edge_max<<<(E + 255) / 256, 256, 0, st>>>(src, dst, S.als, S.ald, S.mEnc, E);
  edge_scatter<<<((size_t)E * 32 + 255) / 256, 256, 0, st>>>(
      src, dst, S.als, S.ald, S.mEnc, S.hs, S.den, S.out, E);
  finalize<<<(n_dst * 64 + 255) / 256, 256, 0, st>>>(S.out, S.den, b, h_out, n_dst);
}

extern "C" void kernel_launch(void* const* d_in, const int* in_sizes, int n_in,
                              void* d_out, int out_size, void* d_ws, size_t ws_size,
                              hipStream_t stream) {
  const float* x_person = (const float*)d_in[0];
  const float* x_product = (const float*)d_in[1];
  const int* pv_src = (const int*)d_in[2];
  const int* pv_dst = (const int*)d_in[3];
  const int* vp_src = (const int*)d_in[4];
  const int* vp_dst = (const int*)d_in[5];
  const float* W_src = (const float*)d_in[6];   // [2,2,64,64]
  const float* W_dst = (const float*)d_in[7];
  const float* a_src = (const float*)d_in[8];   // [2,2,64]
  const float* a_dst = (const float*)d_in[9];
  const float* bias  = (const float*)d_in[10];

  const int NP = in_sizes[0] / 64;
  const int NQ = in_sizes[1] / 64;
  const int E  = in_sizes[2];
  (void)n_in; (void)out_size; (void)ws_size;

  float* ws = (float*)d_ws;
  size_t o = 0;
  float* hpA = ws + o; o += (size_t)NP * 64;     // layer-0 person output
  float* hqA = ws + o; o += (size_t)NQ * 64;     // layer-0 product output
  Scratch S;
  S.hs   = ws + o; o += (size_t)NP * 64;         // src projection (max N)
  S.out  = ws + o; o += (size_t)NP * 64;         // dst accumulator (max N)
  S.als  = ws + o; o += NP;
  S.ald  = ws + o; o += NP;
  S.den  = ws + o; o += NP;
  S.mEnc = (int*)(ws + o); o += NP;
  S.wv   = ws + o; o += 64;

  float* out_hp = (float*)d_out;                 // elu(hp) first (return order)
  float* out_hq = (float*)d_out + (size_t)NP * 64;

  // ---- layer 0 (params [0,0] and [0,1]); both convs read original inputs ----
  run_conv(x_person, x_product, NP, NQ, pv_src, pv_dst, E,
           W_src + 0 * 4096, W_dst + 0 * 4096, a_src + 0 * 64, a_dst + 0 * 64,
           bias + 0 * 64, hqA, S, stream);
  run_conv(x_product, x_person, NQ, NP, vp_src, vp_dst, E,
           W_src + 1 * 4096, W_dst + 1 * 4096, a_src + 1 * 64, a_dst + 1 * 64,
           bias + 1 * 64, hpA, S, stream);

  // ---- layer 1 (params [1,0] and [1,1]); read hpA/hqA, write d_out ----------
  run_conv(hpA, hqA, NP, NQ, pv_src, pv_dst, E,
           W_src + 2 * 4096, W_dst + 2 * 4096, a_src + 2 * 64, a_dst + 2 * 64,
           bias + 2 * 64, out_hq, S, stream);
  run_conv(hqA, hpA, NQ, NP, vp_src, vp_dst, E,
           W_src + 3 * 4096, W_dst + 3 * 4096, a_src + 3 * 64, a_dst + 3 * 64,
           bias + 3 * 64, out_hp, S, stream);
}
